// AMGEdgePolicy_68676527063441
// MI455X (gfx1250) — compile-verified, matrix operand order checked by reading, not run on hardware
//
#include <hip/hip_runtime.h>
#include <hip/hip_bf16.h>

typedef __attribute__((ext_vector_type(16))) _Float16 v16h;
typedef __attribute__((ext_vector_type(8)))  _Float16 v8h;
typedef __attribute__((ext_vector_type(8)))  float    v8f;

#define N_NODES 50000
#define F_IN    16
#define D_HID   128
#define E_EDGES 800000

// 16-bit A/B fragment K index (CDNA5 ISA 7.12.2): lanes 0-15 hold K
// {0..7,16..23}, lanes 16-31 hold K {8..15,24..31}. Each half is a
// contiguous run of 8 halfwords -> exactly two b128 loads per fragment.
__device__ __forceinline__ v16h load_frag16(const _Float16* p) {
  v8h lo = *(const v8h*)(p);
  v8h hi = *(const v8h*)(p + 16);
  return __builtin_shufflevector(lo, hi, 0, 1, 2, 3, 4, 5, 6, 7,
                                 8, 9, 10, 11, 12, 13, 14, 15);
}

__device__ __forceinline__ v8f wmma_f16(v16h a, v16h b, v8f c) {
  return __builtin_amdgcn_wmma_f32_16x16x32_f16(
      /*neg_a=*/false, a, /*neg_b=*/false, b,
      /*c_mod=*/(short)0, c, /*reuse_a=*/false, /*reuse_b=*/false);
}

// ---------------------------------------------------------------- utilities
__global__ void zero_kernel(float* __restrict__ p, int n) {
  int i = blockIdx.x * blockDim.x + threadIdx.x;
  if (i < n) p[i] = 0.0f;
}

__global__ void cvt_f32_to_f16_kernel(const float* __restrict__ src,
                                      _Float16* __restrict__ dst, int n) {
  int i = blockIdx.x * blockDim.x + threadIdx.x;
  if (i < n) dst[i] = (_Float16)src[i];
}

// dst[n*dstStride + dstOff + k] = (f16)src[k*N + n]   -- weight transpose to
// [N][K] so B-fragments are contiguous K-runs (two b128 loads per fragment).
__global__ void transpose_f16_kernel(_Float16* __restrict__ dst,
                                     const float* __restrict__ src,
                                     int K, int N, int dstStride, int dstOff) {
  int idx = blockIdx.x * blockDim.x + threadIdx.x;
  if (idx >= K * N) return;
  int k = idx / N, n = idx - k * N;
  dst[(size_t)n * dstStride + dstOff + k] = (_Float16)src[(size_t)k * N + n];
}

// agg[row[e]][f] += ew[e] * x[col[e]][f]   (F = 16, x fp32)
__global__ void scatter16_kernel(const float* __restrict__ x,
                                 const int* __restrict__ row,
                                 const int* __restrict__ col,
                                 const float* __restrict__ ew,
                                 float* __restrict__ agg) {
  int idx = blockIdx.x * blockDim.x + threadIdx.x;   // E*16 threads exactly
  int e = idx >> 4, f = idx & 15;
  atomicAdd(&agg[row[e] * F_IN + f], ew[e] * x[col[e] * F_IN + f]);
}

// agg[row[e]][f] += ew[e] * h[col[e]][f]   (D = 128, h f16, accum f32)
__global__ void scatter128_kernel(const _Float16* __restrict__ h,
                                  const int* __restrict__ row,
                                  const int* __restrict__ col,
                                  const float* __restrict__ ew,
                                  float* __restrict__ agg) {
  int idx = blockIdx.x * blockDim.x + threadIdx.x;   // E*128 threads exactly
  int e = idx >> 7, f = idx & 127;
  atomicAdd(&agg[row[e] * D_HID + f],
            ew[e] * (float)h[(size_t)col[e] * D_HID + f]);
}

// ------------------------------------------- layer 1: [x|agg] (16x32) @ W (32x128)
// K = 32 == a single v_wmma_f32_16x16x32_f16 per n-tile. Small kernel; scalar
// f32 loads are fine here.
__global__ void gemm1_kernel(const float* __restrict__ x,
                             const float* __restrict__ agg,
                             const float* __restrict__ Ws,
                             const float* __restrict__ Wn,
                             const float* __restrict__ b,
                             _Float16* __restrict__ hout) {
  int lane = threadIdx.x & 31;
  int tile = (blockIdx.x * blockDim.x + threadIdx.x) >> 5;   // wave-uniform
  if (tile >= N_NODES / 16) return;
  int m = lane & 15;
  int off = (lane & 16) ? 8 : 0;
  int node = tile * 16 + m;
  const float* xr = x + (size_t)node * F_IN;
  const float* ar = agg + (size_t)node * F_IN;

  v16h a;
#pragma unroll
  for (int i = 0; i < 16; ++i) {
    int k = off + i + ((i & 8) ? 8 : 0);        // 0..31
    a[i] = (_Float16)(k < 16 ? xr[k] : ar[k - 16]);
  }
#pragma unroll 1
  for (int nt = 0; nt < 8; ++nt) {
    int n = nt * 16 + m;
    float bias = b[n];
    v8f c;
#pragma unroll
    for (int r = 0; r < 8; ++r) c[r] = bias;
    v16h bf;
#pragma unroll
    for (int i = 0; i < 16; ++i) {
      int k = off + i + ((i & 8) ? 8 : 0);
      bf[i] = (_Float16)(k < 16 ? Ws[k * D_HID + n] : Wn[(k - 16) * D_HID + n]);
    }
    c = wmma_f16(a, bf, c);
#pragma unroll
    for (int r = 0; r < 8; ++r) {
      float v = c[r];
      hout[(size_t)(tile * 16 + r + off) * D_HID + n] =
          (_Float16)(v > 0.0f ? v : 0.0f);
    }
  }
}

// ---------------------------- layers 2/3: [h|agg] (16x256) @ Wt (128 cols x 256)
// h, agg in f16; Wt pre-transposed [n][k] with k = 0..255 ([Ws;Wn] stacked).
// All fragment loads are b128 pairs.
__global__ void gemmd_kernel(const _Float16* __restrict__ hin,
                             const _Float16* __restrict__ agg,
                             const _Float16* __restrict__ Wt,
                             const float* __restrict__ b,
                             _Float16* __restrict__ hout) {
  int lane = threadIdx.x & 31;
  int tile = (blockIdx.x * blockDim.x + threadIdx.x) >> 5;   // wave-uniform
  if (tile >= N_NODES / 16) return;
  int m = lane & 15;
  int off = (lane & 16) ? 8 : 0;
  int node = tile * 16 + m;
  const _Float16* hr = hin + (size_t)node * D_HID;
  const _Float16* ar = agg + (size_t)node * D_HID;

  v8f c[8];
#pragma unroll
  for (int nt = 0; nt < 8; ++nt) {
    float bias = b[nt * 16 + m];
#pragma unroll
    for (int r = 0; r < 8; ++r) c[nt][r] = bias;
  }
#pragma unroll 1
  for (int kt = 0; kt < 8; ++kt) {             // K = 256 in steps of 32
    int kb = kt * 32 + off;                    // runs never cross the 128 split
    v16h a = load_frag16(kb < 128 ? hr + kb : ar + (kb - 128));
#pragma unroll
    for (int nt = 0; nt < 8; ++nt) {
      v16h bf = load_frag16(Wt + (size_t)(nt * 16 + m) * 256 + kb);
      c[nt] = wmma_f16(a, bf, c[nt]);
    }
  }
#pragma unroll
  for (int nt = 0; nt < 8; ++nt) {
    int n = nt * 16 + m;
#pragma unroll
    for (int r = 0; r < 8; ++r) {
      float v = c[nt][r];
      hout[(size_t)(tile * 16 + r + off) * D_HID + n] =
          (_Float16)(v > 0.0f ? v : 0.0f);
    }
  }
}

// --------------------------- edge MLP: concat(h[row],h[col],ew) -> 128 -> 64 -> 1
// One wave per 16 edges; h in f16, We1t/We2t pre-transposed f16 [n][k].
__global__ void edge_mlp_kernel(const _Float16* __restrict__ h,
                                const int* __restrict__ row,
                                const int* __restrict__ col,
                                const float* __restrict__ ewg,
                                const _Float16* __restrict__ We1t,  // [128][256]
                                const float* __restrict__ We1,      // for row 256
                                const float* __restrict__ be1,
                                const _Float16* __restrict__ We2t,  // [64][128]
                                const float* __restrict__ be2,
                                const float* __restrict__ We3,
                                const float* __restrict__ be3,
                                float* __restrict__ logits) {
  __shared__ __align__(16) _Float16 lds_t[8][16 * 128];  // relu'd L1 tile/wave
  __shared__ __align__(16) _Float16 lds_u[8][16 * 64];   // relu'd L2 tile/wave
  int lane = threadIdx.x & 31;
  int w = threadIdx.x >> 5;
  int tile = blockIdx.x * 8 + w;            // 50000 tiles, exact
  int m = lane & 15;
  int off = (lane & 16) ? 8 : 0;
  int e0 = tile * 16;

  int nr = row[e0 + m];
  int nc = col[e0 + m];
  const _Float16* hr = h + (size_t)nr * D_HID;
  const _Float16* hc = h + (size_t)nc * D_HID;

  // Hide gather latency: prefetch next tile's rows (global_prefetch_b8).
  if (tile + 1 < E_EDGES / 16) {
    __builtin_prefetch(h + (size_t)row[e0 + 16 + m] * D_HID, 0, 0);
    __builtin_prefetch(h + (size_t)col[e0 + 16 + m] * D_HID, 0, 0);
  }

  // All 8 A-fragments (4 k-steps x {row,col}) live in VGPRs; b128 pairs.
  v16h aR[4], aC[4];
#pragma unroll
  for (int kt = 0; kt < 4; ++kt) {
    aR[kt] = load_frag16(hr + kt * 32 + off);
    aC[kt] = load_frag16(hc + kt * 32 + off);
  }
  float ew8[8];
#pragma unroll
  for (int r = 0; r < 8; ++r) ew8[r] = ewg[e0 + r + off];

  // ----- layer 1: 16x257 @ 257x128, relu -> LDS (f16)
#pragma unroll 1
  for (int nt = 0; nt < 8; ++nt) {
    int n = nt * 16 + m;
    float bias = be1[n];
    v8f c;
#pragma unroll
    for (int r = 0; r < 8; ++r) c[r] = bias;
    const _Float16* wrow = We1t + (size_t)n * 256;
#pragma unroll
    for (int kt = 0; kt < 4; ++kt)
      c = wmma_f16(aR[kt], load_frag16(wrow + kt * 32 + off), c);
#pragma unroll
    for (int kt = 0; kt < 4; ++kt)
      c = wmma_f16(aC[kt], load_frag16(wrow + 128 + kt * 32 + off), c);
    float wlast = We1[256 * 128 + n];          // ew rank-1 row of We1
#pragma unroll
    for (int r = 0; r < 8; ++r) {
      float v = c[r] + ew8[r] * wlast;
      v = v > 0.0f ? v : 0.0f;
      lds_t[w][(r + off) * 128 + n] = (_Float16)v;
    }
  }
  asm volatile("s_wait_dscnt 0" ::: "memory");    // wave-local LDS RAW fence

  // ----- layer 2: 16x128 @ 128x64, relu -> LDS (f16); A frags via ds_load_b128
#pragma unroll 1
  for (int nt = 0; nt < 4; ++nt) {
    int n = nt * 16 + m;
    float bias = be2[n];
    v8f c;
#pragma unroll
    for (int r = 0; r < 8; ++r) c[r] = bias;
    const _Float16* wrow = We2t + (size_t)n * 128;
#pragma unroll
    for (int kt = 0; kt < 4; ++kt) {
      v16h a = load_frag16(&lds_t[w][m * 128 + kt * 32 + off]);
      c = wmma_f16(a, load_frag16(wrow + kt * 32 + off), c);
    }
#pragma unroll
    for (int r = 0; r < 8; ++r) {
      float v = c[r];
      lds_u[w][(r + off) * 64 + n] = (_Float16)(v > 0.0f ? v : 0.0f);
    }
  }
  asm volatile("s_wait_dscnt 0" ::: "memory");

  // ----- layer 3: 64 -> 1 dot product, one edge per lane 0..15
  if (lane < 16) {
    int e = e0 + lane;
    float s = be3[0];
#pragma unroll 1
    for (int j = 0; j < 64; ++j)
      s += (float)lds_u[w][lane * 64 + j] * We3[j];
    if (row[e] == col[e]) s = -1.0e9f;
    logits[e] = s;
  }
}

// -------------------------------- per-node heads (tiny: 1.2 GFLOP, VALU is fine)
__global__ void heads_kernel(const _Float16* __restrict__ h,
                             const float* __restrict__ Wb1, const float* __restrict__ bb1,
                             const float* __restrict__ Wb2, const float* __restrict__ bb2,
                             const float* __restrict__ Wk1, const float* __restrict__ bk1,
                             const float* __restrict__ Wk2, const float* __restrict__ bk2,
                             float* __restrict__ outB, float* __restrict__ outKd,
                             float* __restrict__ outKc, float* __restrict__ outKs) {
  int n = blockIdx.x * blockDim.x + threadIdx.x;
  if (n >= N_NODES) return;
  const _Float16* hr = h + (size_t)n * D_HID;

  float o0 = bb2[0], o1 = bb2[1];
#pragma unroll 1
  for (int j = 0; j < 64; ++j) {
    float s = bb1[j];
#pragma unroll 4
    for (int i = 0; i < 128; ++i) s += (float)hr[i] * Wb1[i * 64 + j];
    s = s > 0.0f ? s : 0.0f;
    o0 += s * Wb2[j * 2 + 0];
    o1 += s * Wb2[j * 2 + 1];
  }
  outB[n * 2 + 0] = o0;
  outB[n * 2 + 1] = o1;

  float ka = bk2[0];
#pragma unroll 1
  for (int j = 0; j < 32; ++j) {
    float s = bk1[j];
#pragma unroll 4
    for (int i = 0; i < 128; ++i) s += (float)hr[i] * Wk1[i * 32 + j];
    s = s > 0.0f ? s : 0.0f;
    ka += s * Wk2[j];
  }
  float kc = 1.0f + 7.0f / (1.0f + __expf(-ka));  // K_MIN + (K_MAX-K_MIN)*sigmoid
  float kr = rintf(kc);                            // round-half-even, matches jnp
  kr = kr < 1.0f ? 1.0f : (kr > 8.0f ? 8.0f : kr);
  ((int*)outKd)[n] = (int)kr;                      // int32 output bits
  outKc[n] = kc;
  outKs[n] = kr;                                   // STE value == k_disc at fwd
}

extern "C" void kernel_launch(void* const* d_in, const int* in_sizes, int n_in,
                              void* d_out, int out_size, void* d_ws, size_t ws_size,
                              hipStream_t stream) {
  const float* x   = (const float*)d_in[0];
  const int*   ei  = (const int*)d_in[1];     // (2,E) int32
  const float* ew  = (const float*)d_in[2];
  const float* Ws1 = (const float*)d_in[3];
  const float* Wn1 = (const float*)d_in[4];
  const float* b1  = (const float*)d_in[5];
  const float* Ws2 = (const float*)d_in[6];
  const float* Wn2 = (const float*)d_in[7];
  const float* b2  = (const float*)d_in[8];
  const float* Ws3 = (const float*)d_in[9];
  const float* Wn3 = (const float*)d_in[10];
  const float* b3  = (const float*)d_in[11];
  const float* We1 = (const float*)d_in[12];
  const float* be1 = (const float*)d_in[13];
  const float* We2 = (const float*)d_in[14];
  const float* be2 = (const float*)d_in[15];
  const float* We3 = (const float*)d_in[16];
  const float* be3 = (const float*)d_in[17];
  const float* Wb1 = (const float*)d_in[18];
  const float* bb1 = (const float*)d_in[19];
  const float* Wb2 = (const float*)d_in[20];
  const float* bb2 = (const float*)d_in[21];
  const float* Wk1 = (const float*)d_in[22];
  const float* bk1 = (const float*)d_in[23];
  const float* Wk2 = (const float*)d_in[24];
  const float* bk2 = (const float*)d_in[25];

  const int* row = ei;
  const int* col = ei + E_EDGES;

  // workspace layout (all offsets 16B-aligned)
  char* p = (char*)d_ws;
  float*    agg0  = (float*)p;     p += (size_t)N_NODES * F_IN  * 4;  // 3.2 MB
  float*    aggD  = (float*)p;     p += (size_t)N_NODES * D_HID * 4;  // 25.6 MB
  _Float16* aggDH = (_Float16*)p;  p += (size_t)N_NODES * D_HID * 2;  // 12.8 MB
  _Float16* hAH   = (_Float16*)p;  p += (size_t)N_NODES * D_HID * 2;
  _Float16* hBH   = (_Float16*)p;  p += (size_t)N_NODES * D_HID * 2;
  _Float16* Wt2   = (_Float16*)p;  p += (size_t)128 * 256 * 2;
  _Float16* Wt3   = (_Float16*)p;  p += (size_t)128 * 256 * 2;
  _Float16* We1t  = (_Float16*)p;  p += (size_t)128 * 256 * 2;
  _Float16* We2t  = (_Float16*)p;  p += (size_t)64 * 128 * 2;

  // output: logits(E) | B_extra(2N) | k_disc(N,int32) | k_cont(N) | k_ste(N)
  float* out    = (float*)d_out;
  float* logits = out;
  float* outB   = out + E_EDGES;
  float* outKd  = outB + 2 * N_NODES;
  float* outKc  = outKd + N_NODES;
  float* outKs  = outKc + N_NODES;

  const int n_tiles = N_NODES / 16;               // 3125, exact
  const int gemm_blocks = (n_tiles + 7) / 8;      // 8 waves / block

  // ---- weight prep: f16 [N][K] transposed/stacked layouts
  transpose_f16_kernel<<<(128 * 128 + 255) / 256, 256, 0, stream>>>(Wt2, Ws2, 128, 128, 256, 0);
  transpose_f16_kernel<<<(128 * 128 + 255) / 256, 256, 0, stream>>>(Wt2, Wn2, 128, 128, 256, 128);
  transpose_f16_kernel<<<(128 * 128 + 255) / 256, 256, 0, stream>>>(Wt3, Ws3, 128, 128, 256, 0);
  transpose_f16_kernel<<<(128 * 128 + 255) / 256, 256, 0, stream>>>(Wt3, Wn3, 128, 128, 256, 128);
  transpose_f16_kernel<<<(256 * 128 + 255) / 256, 256, 0, stream>>>(We1t, We1, 256, 128, 256, 0);
  transpose_f16_kernel<<<(128 * 64 + 255) / 256, 256, 0, stream>>>(We2t, We2, 128, 64, 128, 0);

  // ---- GNN layer 1 (F=16 -> D=128)
  zero_kernel<<<(N_NODES * F_IN + 255) / 256, 256, 0, stream>>>(agg0, N_NODES * F_IN);
  scatter16_kernel<<<E_EDGES * F_IN / 256, 256, 0, stream>>>(x, row, col, ew, agg0);
  gemm1_kernel<<<gemm_blocks, 256, 0, stream>>>(x, agg0, Ws1, Wn1, b1, hAH);

  // ---- GNN layer 2
  zero_kernel<<<(N_NODES * D_HID + 255) / 256, 256, 0, stream>>>(aggD, N_NODES * D_HID);
  scatter128_kernel<<<E_EDGES * D_HID / 256, 256, 0, stream>>>(hAH, row, col, ew, aggD);
  cvt_f32_to_f16_kernel<<<(N_NODES * D_HID + 255) / 256, 256, 0, stream>>>(aggD, aggDH, N_NODES * D_HID);
  gemmd_kernel<<<gemm_blocks, 256, 0, stream>>>(hAH, aggDH, Wt2, b2, hBH);

  // ---- GNN layer 3
  zero_kernel<<<(N_NODES * D_HID + 255) / 256, 256, 0, stream>>>(aggD, N_NODES * D_HID);
  scatter128_kernel<<<E_EDGES * D_HID / 256, 256, 0, stream>>>(hBH, row, col, ew, aggD);
  cvt_f32_to_f16_kernel<<<(N_NODES * D_HID + 255) / 256, 256, 0, stream>>>(aggD, aggDH, N_NODES * D_HID);
  gemmd_kernel<<<gemm_blocks, 256, 0, stream>>>(hBH, aggDH, Wt3, b3, hAH);

  // ---- per-node heads
  heads_kernel<<<(N_NODES + 255) / 256, 256, 0, stream>>>(
      hAH, Wb1, bb1, Wb2, bb2, Wk1, bk1, Wk2, bk2, outB, outKd, outKc, outKs);

  // ---- edge MLP (dominant compute, WMMA)
  edge_mlp_kernel<<<E_EDGES / 16 / 8, 256, 0, stream>>>(
      hAH, row, col, ew, We1t, We1, be1, We2t, be2, We3, be3, logits);
}
